// GNNEncoder_14474039787538
// MI455X (gfx1250) — matrix operand order repaired and checked by reading, not compiled, not run on hardware
//
#include <hip/hip_runtime.h>

// GraphSAGE 2-layer forward for MI455X (gfx1250, wave32).
// Edge phase: build CSR once (degree -> block-scan alloc -> fill), then
//             gather-based mean aggregation (pure coalesced loads, no f32 atomics).
// Dense phase: v_wmma_f32_16x16x32_bf16, weights staged in LDS pre-swizzled to
//              fragment order (conflict-free ds_load), f32 accum, fused bias/ReLU.

typedef __attribute__((ext_vector_type(16))) __bf16 v16bf;
typedef __attribute__((ext_vector_type(8)))  __bf16 v8bf;
typedef __attribute__((ext_vector_type(8)))  float  v8f;

#define D_FEAT 128

__global__ void zero_b32_kernel(unsigned* __restrict__ p, size_t n) {
  size_t i = (size_t)blockIdx.x * blockDim.x + threadIdx.x;
  if (i < n) p[i] = 0u;
}

// W [K][N] f32 row-major  ->  WT [N][K] bf16 row-major
__global__ void transpose_bf16_kernel(const float* __restrict__ W, __bf16* __restrict__ WT,
                                      int K, int N) {
  int i = blockIdx.x * blockDim.x + threadIdx.x;
  if (i >= K * N) return;
  int k = i / N;
  int n = i - k * N;
  WT[n * K + k] = (__bf16)W[i];
}

// ---- CSR construction -------------------------------------------------------

__global__ void degree_kernel(const int* __restrict__ ei, int* __restrict__ deg, int E) {
  int e = blockIdx.x * blockDim.x + threadIdx.x;
  if (e < E) atomicAdd(&deg[ei[E + e]], 1);
}

// Exclusive row offsets: block-level Hillis-Steele scan + one cursor atomic per block.
__global__ void __launch_bounds__(256) alloc_rows_kernel(const int* __restrict__ deg,
                                                         int* __restrict__ rowptr,
                                                         int* __restrict__ cursor, int N) {
  __shared__ int s[256];
  __shared__ int base;
  int tid = threadIdx.x;
  int n = blockIdx.x * 256 + tid;
  int d = (n < N) ? deg[n] : 0;
  s[tid] = d;
  __syncthreads();
#pragma unroll
  for (int off = 1; off < 256; off <<= 1) {
    int v = (tid >= off) ? s[tid - off] : 0;
    __syncthreads();
    s[tid] += v;
    __syncthreads();
  }
  if (tid == 255) base = atomicAdd(cursor, s[255]);
  __syncthreads();
  if (n < N) rowptr[n] = base + s[tid] - d;  // exclusive within block + global base
}

__global__ void fill_adj_kernel(const int* __restrict__ ei, const int* __restrict__ rowptr,
                                int* __restrict__ fill, int* __restrict__ adj, int E) {
  int e = blockIdx.x * blockDim.x + threadIdx.x;
  if (e >= E) return;
  int src = ei[e];
  int dst = ei[E + e];
  int pos = atomicAdd(&fill[dst], 1);
  adj[rowptr[dst] + pos] = src;
}

// ---- Gather-based mean aggregation (wave32 per node, fused mean) ------------

__global__ void gather_mean_kernel(const float* __restrict__ feat, const int* __restrict__ adj,
                                   const int* __restrict__ rowptr, const int* __restrict__ deg,
                                   float* __restrict__ agg, int N) {
  int wid  = (int)(((size_t)blockIdx.x * blockDim.x + threadIdx.x) >> 5);
  int lane = threadIdx.x & 31;
  if (wid >= N) return;
  int start = rowptr[wid];   // wave-uniform -> scalar loads
  int d     = deg[wid];
  const int off = lane * 4;
  float a0 = 0.f, a1 = 0.f, a2 = 0.f, a3 = 0.f;
  float b0 = 0.f, b1 = 0.f, b2 = 0.f, b3 = 0.f;
  int j = 0;
  for (; j + 2 <= d; j += 2) {
    const float4 u = *(const float4*)(feat + (size_t)adj[start + j] * D_FEAT + off);
    const float4 v = *(const float4*)(feat + (size_t)adj[start + j + 1] * D_FEAT + off);
    a0 += u.x; a1 += u.y; a2 += u.z; a3 += u.w;
    b0 += v.x; b1 += v.y; b2 += v.z; b3 += v.w;
  }
  if (j < d) {
    const float4 u = *(const float4*)(feat + (size_t)adj[start + j] * D_FEAT + off);
    a0 += u.x; a1 += u.y; a2 += u.z; a3 += u.w;
  }
  float inv = 1.0f / (float)(d > 0 ? d : 1);
  float4 r;
  r.x = (a0 + b0) * inv;
  r.y = (a1 + b1) * inv;
  r.z = (a2 + b2) * inv;
  r.w = (a3 + b3) * inv;
  *(float4*)(agg + (size_t)wid * D_FEAT + off) = r;
}

// ---- WMMA GEMM --------------------------------------------------------------

// A fragment (16x32 bf16, wave32): lane holds row M = lane&15.
// Element j -> k = (j&7) + 16*(j>>3) + 8*half  => two contiguous 8-float chunks.
__device__ __forceinline__ v16bf load_a_frag(const float* __restrict__ row, int kb, int half) {
  const float* p0 = row + kb + 8 * half;
  const float* p1 = p0 + 16;
  v16bf r;
#pragma unroll
  for (int j = 0; j < 8; ++j) r[j] = (__bf16)p0[j];
#pragma unroll
  for (int j = 0; j < 8; ++j) r[8 + j] = (__bf16)p1[j];
  return r;
}

__device__ __forceinline__ v16bf load_b_lds(const __bf16* p) {
  v8bf lo = *(const v8bf*)p;        // 16B ds_load
  v8bf hi = *(const v8bf*)(p + 8);  // 16B ds_load
  v16bf r;
#pragma unroll
  for (int j = 0; j < 8; ++j) { r[j] = lo[j]; r[j + 8] = hi[j]; }
  return r;
}

// out[16 rows x NCOLS] per wave:  out = A1@W1 + A2@W2 + bias  (optional ReLU)
// Weights (bf16, transposed [NCOLS][128]) are staged into LDS in fragment order:
// sW[frag][lane][16 elems], frag = (which*NT + t)*KB + kb32, so each B fragment
// read is lane-contiguous (32B/lane) and bank-conflict-free.
template <int NCOLS, bool RELU>
__global__ void __launch_bounds__(256) sage_gemm_kernel(
    const float* __restrict__ A1, const float* __restrict__ A2,
    const __bf16* __restrict__ W1T, const __bf16* __restrict__ W2T,
    const float* __restrict__ bias, float* __restrict__ out, int nrows) {
  constexpr int K     = 128;
  constexpr int NT    = NCOLS / 16;
  constexpr int KB    = K / 32;        // k-steps
  constexpr int FRAGS = NT * KB;       // fragments per matrix
  __shared__ __bf16 sW[2 * FRAGS * 512];  // 512 bf16 per fragment

  // Cooperative swizzled staging: 16B chunks. 64 chunks per fragment.
  for (int c = threadIdx.x; c < 2 * FRAGS * 64; c += (int)blockDim.x) {
    int which = (c >= FRAGS * 64) ? 1 : 0;
    int cc    = c - which * FRAGS * 64;
    int frag  = cc >> 6;
    int l     = (cc >> 1) & 31;
    int jh    = cc & 1;
    int t     = frag / KB;
    int kb32  = frag - t * KB;
    const __bf16* Wt  = which ? W2T : W1T;
    // B layout: lane l -> column n = l&15, k = kb32*32 + 8*(l>>4) + (j&7) + 16*(j>>3)
    const __bf16* src = Wt + (t * 16 + (l & 15)) * K + kb32 * 32 + (l >> 4) * 8 + jh * 16;
    __bf16* dst = sW + ((size_t)(which * FRAGS + frag) * 32 + l) * 16 + jh * 8;
    *(ulonglong2*)dst = *(const ulonglong2*)src;
  }
  __syncthreads();

  int wid     = blockIdx.x * ((int)blockDim.x >> 5) + ((int)threadIdx.x >> 5);
  int lane    = threadIdx.x & 31;
  int half    = lane >> 4;
  int m16     = lane & 15;
  int rowbase = wid * 16;
  if (rowbase >= nrows) return;  // wave-uniform: EXEC all-1s for WMMA

  v8f acc[NT] = {};
  const float* rowA1 = A1 + (size_t)(rowbase + m16) * K;
  const float* rowA2 = A2 + (size_t)(rowbase + m16) * K;

#pragma unroll
  for (int kb32 = 0; kb32 < KB; ++kb32) {
    v16bf a1 = load_a_frag(rowA1, kb32 * 32, half);
    v16bf a2 = load_a_frag(rowA2, kb32 * 32, half);
#pragma unroll
    for (int t = 0; t < NT; ++t) {
      const __bf16* f1 = sW + ((size_t)(t * KB + kb32) * 32 + lane) * 16;
      const __bf16* f2 = sW + ((size_t)((FRAGS + t * KB + kb32)) * 32 + lane) * 16;
      v16bf b1 = load_b_lds(f1);
      v16bf b2 = load_b_lds(f2);
      acc[t] = __builtin_amdgcn_wmma_f32_16x16x32_bf16(false, a1, false, b1,
                                                       (short)0, acc[t], false, false);
      acc[t] = __builtin_amdgcn_wmma_f32_16x16x32_bf16(false, a2, false, b2,
                                                       (short)0, acc[t], false, false);
    }
  }

  // C/D layout: VGPR r, lane -> row = r + 8*half, col = lane&15
#pragma unroll
  for (int t = 0; t < NT; ++t) {
    int col  = t * 16 + m16;
    float bv = bias[col];
#pragma unroll
    for (int r = 0; r < 8; ++r) {
      float v = acc[t][r] + bv;
      if (RELU) v = fmaxf(v, 0.0f);
      out[(size_t)(rowbase + r + 8 * half) * NCOLS + col] = v;
    }
  }
}

// ---- Host-side orchestration ------------------------------------------------

extern "C" void kernel_launch(void* const* d_in, const int* in_sizes, int n_in,
                              void* d_out, int out_size, void* d_ws, size_t ws_size,
                              hipStream_t stream) {
  const float* x   = (const float*)d_in[0];
  const int*   ei  = (const int*)d_in[1];   // [2][E]
  const float* Wl1 = (const float*)d_in[2];
  const float* Wr1 = (const float*)d_in[3];
  const float* b1  = (const float*)d_in[4];
  const float* Wl2 = (const float*)d_in[5];
  const float* Wr2 = (const float*)d_in[6];
  const float* b2  = (const float*)d_in[7];
  float* out = (float*)d_out;

  const int N = in_sizes[0] / D_FEAT;  // 100000
  const int E = in_sizes[1] / 2;       // 1600000

  // Workspace layout (16B alignment preserved at every section boundary):
  float* agg    = (float*)d_ws;                 // N*128 f32
  float* h      = agg + (size_t)N * D_FEAT;     // N*128 f32
  int*   deg    = (int*)(h + (size_t)N * D_FEAT);  // N
  int*   fill   = deg + N;                      // N
  int*   cursor = fill + N;                     // 16 ints (1 used, pad for alignment)
  int*   rowptr = cursor + 16;                  // N
  int*   adj    = rowptr + N;                   // E
  __bf16* W1lT  = (__bf16*)(adj + E);           // 128*128 bf16
  __bf16* W1rT  = W1lT + 128 * 128;             // 128*128 bf16
  __bf16* W2lT  = W1rT + 128 * 128;             // 64*128 bf16
  __bf16* W2rT  = W2lT + 64 * 128;              // 64*128 bf16

  // Pre-transpose weights to bf16 [N][K]
  transpose_bf16_kernel<<<(128 * 128 + 255) / 256, 256, 0, stream>>>(Wl1, W1lT, 128, 128);
  transpose_bf16_kernel<<<(128 * 128 + 255) / 256, 256, 0, stream>>>(Wr1, W1rT, 128, 128);
  transpose_bf16_kernel<<<(128 * 64 + 255) / 256, 256, 0, stream>>>(Wl2, W2lT, 128, 64);
  transpose_bf16_kernel<<<(128 * 64 + 255) / 256, 256, 0, stream>>>(Wr2, W2rT, 128, 64);

  // Build CSR (shared by both layers)
  size_t zcnt = (size_t)2 * N + 16;  // deg + fill + cursor
  zero_b32_kernel<<<(int)((zcnt + 255) / 256), 256, 0, stream>>>((unsigned*)deg, zcnt);
  degree_kernel<<<(E + 255) / 256, 256, 0, stream>>>(ei, deg, E);
  alloc_rows_kernel<<<(N + 255) / 256, 256, 0, stream>>>(deg, rowptr, cursor, N);
  fill_adj_kernel<<<(E + 255) / 256, 256, 0, stream>>>(ei, rowptr, fill, adj, E);

  const int gatherBlocks = (int)(((size_t)N * 32 + 255) / 256);
  const int tiles   = (N + 15) / 16;
  const int gblocks = (tiles + 7) / 8;  // 8 waves / block

  // ---- Layer 1 ----
  gather_mean_kernel<<<gatherBlocks, 256, 0, stream>>>(x, adj, rowptr, deg, agg, N);
  sage_gemm_kernel<128, true><<<gblocks, 256, 0, stream>>>(agg, x, W1lT, W1rT, b1, h, N);

  // ---- Layer 2 ----
  gather_mean_kernel<<<gatherBlocks, 256, 0, stream>>>(h, adj, rowptr, deg, agg, N);
  sage_gemm_kernel<64, false><<<gblocks, 256, 0, stream>>>(agg, h, W2lT, W2rT, b2, out, N);
}
